// CameraAwareSparseBlock_40458591928948
// MI455X (gfx1250) — matrix operand order repaired and checked by reading, not compiled, not run on hardware
//
#include <hip/hip_runtime.h>

#define NV   200000
#define K27  27
#define CIN  32
#define COUT 64
#define CAMD 256
#define EPSV 1e-5f

typedef __attribute__((ext_vector_type(16))) _Float16 v16h;
typedef __attribute__((ext_vector_type(8)))  _Float16 h8;
typedef __attribute__((ext_vector_type(8)))  float    v8f;

union AU { h8 h[2]; v16h v; };

__device__ __forceinline__ v8f wmma16(v16h a, v16h b, v8f c) {
  return __builtin_amdgcn_wmma_f32_16x16x32_f16(false, a, false, b, (short)0, c, false, false);
}

// idx<0 -> zero row stored at index NV
__device__ __forceinline__ int zsel(int i) { return i < 0 ? NV : i; }

// ---------------- small prep kernels ----------------

// converts nvalid elements, zero-fills up to ntot (the sentinel zero row)
__global__ void cvt_feats_k(const float* __restrict__ f, _Float16* __restrict__ o,
                            int nvalid, int ntot) {
  int i = blockIdx.x * 256 + threadIdx.x;
  if (i < ntot) o[i] = (i < nvalid) ? (_Float16)f[i] : (_Float16)0.f;
}

// dst[k][j][i] = (f16) src[k][i][j]  (transpose per-tap weight to col-major for B-tiles)
__global__ void cvt_w_k(const float* __restrict__ w, _Float16* __restrict__ o,
                        int KK, int Cin, int Cout) {
  int i = blockIdx.x * 256 + threadIdx.x;
  int tot = KK * Cin * Cout;
  if (i >= tot) return;
  int k   = i / (Cin * Cout);
  int rem = i - k * Cin * Cout;
  int j   = rem / Cin;        // out channel
  int ii  = rem - j * Cin;    // in  channel
  o[i] = (_Float16)w[(k * Cin + ii) * Cout + j];
}

// cam[b][0..63]=scale, cam[b][64..127]=shift : cond(4x256) @ Wc(256x128) + bc
__global__ void cam_k(const float* __restrict__ cond, const float* __restrict__ Wc,
                      const float* __restrict__ bc, float* __restrict__ cam) {
  int t = threadIdx.x;            // 512 threads
  int b = t >> 7, j = t & 127;
  float s = bc[j];
  for (int i = 0; i < CAMD; i++) s += cond[b * CAMD + i] * Wc[i * 128 + j];
  cam[t] = s;
}

// ---------------- WMMA conv kernels (LDS weights + pipelined gathers) ----------------

__global__ void __launch_bounds__(256)
conv1_k(const _Float16* __restrict__ featsH, const _Float16* __restrict__ W1T,
        const int* __restrict__ nbr, const float* __restrict__ b1,
        float* __restrict__ h1) {
  extern __shared__ __align__(16) char smemraw[];
  _Float16* sW = (_Float16*)smemraw;
  {   // cooperative stage of all 27 taps: 27*64*32 f16 = 108 KB
    const float4* src = (const float4*)W1T;
    float4*       dst = (float4*)smemraw;
    for (int j = threadIdx.x; j < (K27 * COUT * CIN) / 8; j += 256) dst[j] = src[j];
  }
  __syncthreads();
  const int lane = threadIdx.x & 31;
  const int wave = threadIdx.x >> 5;
  const int tile = blockIdx.x * 8 + wave;          // 32-row tiles, 6250 total
  if (tile < NV / 32) {                            // wave-uniform guard: EXEC all-ones inside
    const int half = lane >> 4;
    const int colL = lane & 15;
    const int n0   = tile * 32;
    const int* pA  = nbr + (n0 + colL) * K27;
    const int* pB  = nbr + (n0 + 16 + colL) * K27;

    // pipeline prologue: idx for taps 0,1; gather tap 0
    int iA1 = pA[1], iB1 = pB[1];
    int jA  = zsel(pA[0]), jB = zsel(pB[0]);
    const _Float16* fA = featsH + jA * CIN + half * 16;
    const _Float16* fB = featsH + jB * CIN + half * 16;
    h8 ca0 = *(const h8*)(fA), ca1 = *(const h8*)(fA + 8);
    h8 cb0 = *(const h8*)(fB), cb1 = *(const h8*)(fB + 8);

    v8f ac[8] = {{}, {}, {}, {}, {}, {}, {}, {}};
    for (int k = 0; k < K27; k++) {
      int kk  = (k + 2 < K27) ? k + 2 : K27 - 1;
      int iA2 = pA[kk], iB2 = pB[kk];              // idx prefetch (2 ahead)
      jA = zsel(iA1);  jB = zsel(iB1);             // gather prefetch (1 ahead)
      const _Float16* gA = featsH + jA * CIN + half * 16;
      const _Float16* gB = featsH + jB * CIN + half * 16;
      h8 na0 = *(const h8*)(gA), na1 = *(const h8*)(gA + 8);
      h8 nb0 = *(const h8*)(gB), nb1 = *(const h8*)(gB + 8);

      AU A1u; A1u.h[0] = ca0; A1u.h[1] = ca1;
      AU A2u; A2u.h[0] = cb0; A2u.h[1] = cb1;
      const _Float16* wp = sW + (k * COUT + colL) * CIN + half * 16;
      v16h w0 = *(const v16h*)(wp);
      v16h w1 = *(const v16h*)(wp + 16 * CIN);
      v16h w2 = *(const v16h*)(wp + 32 * CIN);
      v16h w3 = *(const v16h*)(wp + 48 * CIN);
      ac[0] = wmma16(A1u.v, w0, ac[0]);  ac[4] = wmma16(A2u.v, w0, ac[4]);
      ac[1] = wmma16(A1u.v, w1, ac[1]);  ac[5] = wmma16(A2u.v, w1, ac[5]);
      ac[2] = wmma16(A1u.v, w2, ac[2]);  ac[6] = wmma16(A2u.v, w2, ac[6]);
      ac[3] = wmma16(A1u.v, w3, ac[3]);  ac[7] = wmma16(A2u.v, w3, ac[7]);

      ca0 = na0; ca1 = na1; cb0 = nb0; cb1 = nb1;
      iA1 = iA2; iB1 = iB2;
    }
    float bb0 = b1[colL], bb1 = b1[16 + colL], bb2 = b1[32 + colL], bb3 = b1[48 + colL];
#pragma unroll
    for (int g = 0; g < 2; g++) {
#pragma unroll
      for (int r = 0; r < 8; r++) {
        float* op = h1 + (n0 + g * 16 + r + half * 8) * COUT;
        op[colL]      = ac[g * 4 + 0][r] + bb0;
        op[16 + colL] = ac[g * 4 + 1][r] + bb1;
        op[32 + colL] = ac[g * 4 + 2][r] + bb2;
        op[48 + colL] = ac[g * 4 + 3][r] + bb3;
      }
    }
  }
}

__device__ __forceinline__ void conv2_taps(const _Float16* __restrict__ a1H,
                                           const int* __restrict__ pA,
                                           const int* __restrict__ pB,
                                           const _Float16* sW, int kbeg, int kend,
                                           int half, int colL, v8f ac[8]) {
  // prologue for this pass
  int iA1 = pA[kbeg + 1], iB1 = pB[kbeg + 1];
  int jA  = zsel(pA[kbeg]), jB = zsel(pB[kbeg]);
  const _Float16* fA = a1H + jA * COUT + half * 16;
  const _Float16* fB = a1H + jB * COUT + half * 16;
  h8 cp0 = *(const h8*)(fA),      cp1 = *(const h8*)(fA + 8);
  h8 cp2 = *(const h8*)(fA + 32), cp3 = *(const h8*)(fA + 40);
  h8 cq0 = *(const h8*)(fB),      cq1 = *(const h8*)(fB + 8);
  h8 cq2 = *(const h8*)(fB + 32), cq3 = *(const h8*)(fB + 40);

  for (int k = kbeg; k < kend; k++) {
    int kk  = (k + 2 < kend) ? k + 2 : kend - 1;
    int iA2 = pA[kk], iB2 = pB[kk];
    jA = zsel(iA1);  jB = zsel(iB1);
    const _Float16* gA = a1H + jA * COUT + half * 16;
    const _Float16* gB = a1H + jB * COUT + half * 16;
    h8 np0 = *(const h8*)(gA),      np1 = *(const h8*)(gA + 8);
    h8 np2 = *(const h8*)(gA + 32), np3 = *(const h8*)(gA + 40);
    h8 nq0 = *(const h8*)(gB),      nq1 = *(const h8*)(gB + 8);
    h8 nq2 = *(const h8*)(gB + 32), nq3 = *(const h8*)(gB + 40);

    AU A0; A0.h[0] = cp0; A0.h[1] = cp1;   // rows A, K 0..31
    AU A1; A1.h[0] = cp2; A1.h[1] = cp3;   // rows A, K 32..63
    AU B0; B0.h[0] = cq0; B0.h[1] = cq1;   // rows B, K 0..31
    AU B1; B1.h[0] = cq2; B1.h[1] = cq3;   // rows B, K 32..63
    const _Float16* wp = sW + ((k - kbeg) * COUT + colL) * COUT + half * 16;
#pragma unroll
    for (int c = 0; c < 4; c++) {
      v16h w0 = *(const v16h*)(wp + c * 16 * COUT);       // K 0..31 slice
      v16h w1 = *(const v16h*)(wp + c * 16 * COUT + 32);  // K 32..63 slice
      ac[c]     = wmma16(A0.v, w0, ac[c]);
      ac[c]     = wmma16(A1.v, w1, ac[c]);
      ac[4 + c] = wmma16(B0.v, w0, ac[4 + c]);
      ac[4 + c] = wmma16(B1.v, w1, ac[4 + c]);
    }
    cp0 = np0; cp1 = np1; cp2 = np2; cp3 = np3;
    cq0 = nq0; cq1 = nq1; cq2 = nq2; cq3 = nq3;
    iA1 = iA2; iB1 = iB2;
  }
}

__global__ void __launch_bounds__(256)
conv2_k(const _Float16* __restrict__ a1H, const _Float16* __restrict__ W2T,
        const int* __restrict__ nbr, const float* __restrict__ b2,
        float* __restrict__ h2) {
  extern __shared__ __align__(16) char smemraw[];
  _Float16* sW = (_Float16*)smemraw;
  const int lane = threadIdx.x & 31;
  const int wave = threadIdx.x >> 5;
  const int tile = blockIdx.x * 8 + wave;
  const bool active = (tile < NV / 32);            // wave-uniform
  const int half = lane >> 4;
  const int colL = lane & 15;
  const int n0   = active ? tile * 32 : 0;
  const int* pA  = nbr + (n0 + colL) * K27;
  const int* pB  = nbr + (n0 + 16 + colL) * K27;
  v8f ac[8] = {{}, {}, {}, {}, {}, {}, {}, {}};

  // ---- pass 1: taps 0..12 (13*64*64 f16 = 104 KB) ----
  {
    const float4* src = (const float4*)W2T;
    float4*       dst = (float4*)smemraw;
    for (int j = threadIdx.x; j < (13 * COUT * COUT) / 8; j += 256) dst[j] = src[j];
  }
  __syncthreads();
  if (active) conv2_taps(a1H, pA, pB, sW, 0, 13, half, colL, ac);
  __syncthreads();
  // ---- pass 2: taps 13..26 (14*64*64 f16 = 112 KB) ----
  {
    const float4* src = (const float4*)W2T + (13 * COUT * COUT) / 8;
    float4*       dst = (float4*)smemraw;
    for (int j = threadIdx.x; j < (14 * COUT * COUT) / 8; j += 256) dst[j] = src[j];
  }
  __syncthreads();
  if (active) {
    conv2_taps(a1H, pA, pB, sW, 13, 27, half, colL, ac);
    float bb0 = b2[colL], bb1 = b2[16 + colL], bb2 = b2[32 + colL], bb3 = b2[48 + colL];
#pragma unroll
    for (int g = 0; g < 2; g++) {
#pragma unroll
      for (int r = 0; r < 8; r++) {
        float* op = h2 + (n0 + g * 16 + r + half * 8) * COUT;
        op[colL]      = ac[g * 4 + 0][r] + bb0;
        op[16 + colL] = ac[g * 4 + 1][r] + bb1;
        op[32 + colL] = ac[g * 4 + 2][r] + bb2;
        op[48 + colL] = ac[g * 4 + 3][r] + bb3;
      }
    }
  }
}

// ---------------- deterministic BN statistics ----------------

__global__ void stats_k(const float* __restrict__ h, float* __restrict__ partials) {
  __shared__ float ls[256], lq[256];
  int t = threadIdx.x;
  int c = t & 63;
  float s = 0.f, q = 0.f;
  for (int r = blockIdx.x * 4 + (t >> 6); r < NV; r += 512) {
    float v = h[r * 64 + c];
    s += v; q += v * v;
  }
  ls[t] = s; lq[t] = q;
  __syncthreads();
  if (t < 64) {
    float ss = ls[t] + ls[t + 64] + ls[t + 128] + ls[t + 192];
    float qq = lq[t] + lq[t + 64] + lq[t + 128] + lq[t + 192];
    partials[blockIdx.x * 128 + t]      = ss;
    partials[blockIdx.x * 128 + 64 + t] = qq;
  }
}

__global__ void finalize_k(const float* __restrict__ P, float* __restrict__ mu,
                           float* __restrict__ rstd) {
  int c = threadIdx.x;   // 64 threads
  float s = 0.f, q = 0.f;
  for (int p = 0; p < 128; p++) { s += P[p * 128 + c]; q += P[p * 128 + 64 + c]; }
  float m   = s / (float)NV;
  float var = q / (float)NV - m * m;
  mu[c]   = m;
  rstd[c] = rsqrtf(var + EPSV);
}

// ---------------- elementwise fusions ----------------

__global__ void bnrelu_k(const float* __restrict__ h1, const float* __restrict__ mu,
                         const float* __restrict__ rstd, const float* __restrict__ g,
                         const float* __restrict__ be, _Float16* __restrict__ o,
                         int nvalid, int ntot) {
  int i = blockIdx.x * 256 + threadIdx.x;
  if (i >= ntot) return;
  if (i >= nvalid) { o[i] = (_Float16)0.f; return; }   // sentinel zero row
  int c = i & 63;
  float v = (h1[i] - mu[c]) * rstd[c] * g[c] + be[c];
  o[i] = (_Float16)fmaxf(v, 0.f);
}

__global__ void final_k(float* __restrict__ out, const float* __restrict__ feats,
                        const float* __restrict__ Wr, const float* __restrict__ br,
                        const float* __restrict__ g2, const float* __restrict__ be2,
                        const float* __restrict__ mu2, const float* __restrict__ rstd2,
                        const float* __restrict__ cam, const int* __restrict__ bidx) {
  int i = blockIdx.x * 256 + threadIdx.x;
  if (i >= NV * COUT) return;
  int nrow = i >> 6, c = i & 63;
  float y = (out[i] - mu2[c]) * rstd2[c] * g2[c] + be2[c];
  int b = bidx[nrow];
  y = y * (1.f + cam[b * 128 + c]) + cam[b * 128 + 64 + c];
  y = fmaxf(y, 0.f);
  float id = br[c];
  const float* fr = feats + nrow * CIN;
#pragma unroll
  for (int r = 0; r < CIN; r++) id += fr[r] * Wr[r * COUT + c];
  out[i] = y + id;
}

// ---------------- launch ----------------

extern "C" void kernel_launch(void* const* d_in, const int* in_sizes, int n_in,
                              void* d_out, int out_size, void* d_ws, size_t ws_size,
                              hipStream_t stream) {
  const float* feats = (const float*)d_in[0];
  const float* cond  = (const float*)d_in[1];
  const float* W1    = (const float*)d_in[2];
  const float* b1    = (const float*)d_in[3];
  const float* g1    = (const float*)d_in[4];
  const float* be1   = (const float*)d_in[5];
  const float* W2    = (const float*)d_in[6];
  const float* b2    = (const float*)d_in[7];
  const float* g2    = (const float*)d_in[8];
  const float* be2   = (const float*)d_in[9];
  const float* Wc    = (const float*)d_in[10];
  const float* bc    = (const float*)d_in[11];
  const float* Wr    = (const float*)d_in[12];
  const float* br    = (const float*)d_in[13];
  const int*   nbr   = (const int*)d_in[14];
  const int*   bidx  = (const int*)d_in[15];
  float* out = (float*)d_out;
  char*  ws  = (char*)d_ws;

  _Float16* featsH = (_Float16*)(ws);                 // (N+1)*32 f16 (row N = zeros)
  _Float16* a1H    = (_Float16*)(ws + 12800256);      // (N+1)*64 f16 (row N = zeros)
  float*    h1     = (float*)   (ws + 38400512);      // N*64 f32
  _Float16* W1T    = (_Float16*)(ws + 89600512);      // 27*64*32 f16
  _Float16* W2T    = (_Float16*)(ws + 89711104);      // 27*64*64 f16
  float*    cam    = (float*)   (ws + 89932288);      // 4*128 f32
  float*    parts  = (float*)   (ws + 89932800);      // 128*128 f32
  float*    mu1    = (float*)   (ws + 89998336);
  float*    rstd1  = mu1 + 64;
  float*    mu2    = mu1 + 128;
  float*    rstd2  = mu1 + 192;

  cvt_feats_k<<<((NV + 1) * CIN + 255) / 256, 256, 0, stream>>>(feats, featsH,
                                                                NV * CIN, (NV + 1) * CIN);
  cvt_w_k<<<(K27 * CIN * COUT + 255) / 256, 256, 0, stream>>>(W1, W1T, K27, CIN, COUT);
  cvt_w_k<<<(K27 * COUT * COUT + 255) / 256, 256, 0, stream>>>(W2, W2T, K27, COUT, COUT);
  cam_k<<<1, 512, 0, stream>>>(cond, Wc, bc, cam);

  const int wtiles = NV / 32;                   // 6250 (32-row tiles)
  const int blocks = (wtiles + 7) / 8;          // 782
  conv1_k<<<blocks, 256, K27 * COUT * CIN * 2, stream>>>(featsH, W1T, nbr, b1, h1);
  stats_k<<<128, 256, 0, stream>>>(h1, parts);
  finalize_k<<<1, 64, 0, stream>>>(parts, mu1, rstd1);
  bnrelu_k<<<((NV + 1) * COUT + 255) / 256, 256, 0, stream>>>(h1, mu1, rstd1, g1, be1, a1H,
                                                              NV * COUT, (NV + 1) * COUT);

  conv2_k<<<blocks, 256, 14 * COUT * COUT * 2, stream>>>(a1H, W2T, nbr, b2, out);
  stats_k<<<128, 256, 0, stream>>>(out, parts);
  finalize_k<<<1, 64, 0, stream>>>(parts, mu2, rstd2);
  final_k<<<(NV * COUT + 255) / 256, 256, 0, stream>>>(out, feats, Wr, br, g2, be2,
                                                       mu2, rstd2, cam, bidx);
}
